// Mamba_24266565222485
// MI455X (gfx1250) — matrix operand order repaired and checked by reading, not compiled, not run on hardware
//
#include <hip/hip_runtime.h>
#include <hip/hip_bf16.h>

// ---------------------------------------------------------------------------
// Mamba forward (B=1, L=2048, D_MODEL=768, D_INNER=1536, D_STATE=16,
// DT_RANK=48, D_CONV=4, N_LAYER=4, VOCAB=50264) for gfx1250 (MI455X).
// Dense projections via V_WMMA_F32_16X16X32_BF16 (fp32 accumulate).
// All GEMM operands pre-converted to bf16 (once per call for weights/emb,
// fused into producers for activations) -> cvt-free, hazard-free WMMA loop.
// ---------------------------------------------------------------------------

#define L_SEQ    2048
#define DMODEL   768
#define DINNER   1536
#define DSTATE   16
#define DTRANK   48
#define DCONV    4
#define NLAYER   4
#define NVOCAB   50264
#define XDBL_W   (DTRANK + 2 * DSTATE)   // 80

typedef __attribute__((ext_vector_type(8)))  float  v8f;
typedef __attribute__((ext_vector_type(16))) __bf16 bf16x16;
typedef __attribute__((ext_vector_type(8)))  __bf16 bf16x8;
typedef __attribute__((ext_vector_type(4)))  __bf16 bf16x4;

// ---------------------------------------------------------------------------
// GEMM: C[M,N] = A[M,K] * W[N,K]^T  (+ C if accumulate);  A, W are bf16.
// Block = 128 threads = 4 waves.  Wave tile = 32(M) x 64(N):
//   2 A fragments + 4 B fragments (12 x b128 loads, zero VALU converts),
//   then 8 back-to-back WMMAs per 32-wide K step.
// Block tile = 128(M) x 64(N).  K tail (K % 32) peeled out of the main loop.
// Fragment layouts per CDNA5 ISA 7.12.2:
//   A 16x32 bf16: lane L holds row M=L%16; K = (e&7) + 8*(L/16) + 16*(e/8)
//   B 32x16 bf16: lane L holds col N=L%16; K = (e&7) + 8*(L/16) + 16*(e/8)
//   C 16x16 f32 : lane L holds col N=L%16; row = r + 8*(L/16), r=0..7
// ---------------------------------------------------------------------------
__global__ __launch_bounds__(128) void gemm_bf16_wmma(
    const __bf16* __restrict__ A, int lda,
    const __bf16* __restrict__ W, int ldw,
    float* __restrict__ C, int ldc,
    int M, int N, int K, int accumulate)
{
    const int lane  = threadIdx.x & 31;
    const int wave  = threadIdx.x >> 5;
    const int half  = lane >> 4;        // 0 / 1
    const int lan15 = lane & 15;

    const int m0   = blockIdx.x * 128 + wave * 32;   // wave owns rows m0..m0+31
    const int nblk = blockIdx.y * 64;
    if (m0 >= M) return;

    v8f acc[2][4];
    #pragma unroll
    for (int s = 0; s < 2; ++s)
        #pragma unroll
        for (int t = 0; t < 4; ++t)
            #pragma unroll
            for (int i = 0; i < 8; ++i) acc[s][t][i] = 0.0f;

    const int rA0 = (m0 + lan15 < M)      ? (m0 + lan15)      : (M - 1);
    const int rA1 = (m0 + 16 + lan15 < M) ? (m0 + 16 + lan15) : (M - 1);
    const __bf16* __restrict__ Arow0 = A + (size_t)rA0 * lda;
    const __bf16* __restrict__ Arow1 = A + (size_t)rA1 * lda;

    const __bf16* __restrict__ Wrow[4];
    #pragma unroll
    for (int t = 0; t < 4; ++t) {
        int n = nblk + t * 16 + lan15;
        Wrow[t] = W + (size_t)((n < N) ? n : (N - 1)) * ldw;
    }

    const int Kfull = K & ~31;

    // -------- branch-free, cvt-free main K loop --------
    for (int kk = 0; kk < Kfull; kk += 32) {
        const int kA = kk + half * 8;
        const int kB = kk + half * 8;

        bf16x8 a0lo = *(const bf16x8*)(Arow0 + kA);
        bf16x8 a0hi = *(const bf16x8*)(Arow0 + kA + 16);
        bf16x8 a1lo = *(const bf16x8*)(Arow1 + kA);
        bf16x8 a1hi = *(const bf16x8*)(Arow1 + kA + 16);
        bf16x16 a0 = __builtin_shufflevector(a0lo, a0hi,
            0,1,2,3,4,5,6,7,8,9,10,11,12,13,14,15);
        bf16x16 a1 = __builtin_shufflevector(a1lo, a1hi,
            0,1,2,3,4,5,6,7,8,9,10,11,12,13,14,15);

        bf16x16 b[4];
        #pragma unroll
        for (int t = 0; t < 4; ++t) {
            __builtin_prefetch(Wrow[t] + kB + 64, 0, 3);  // speculative; drop-on-fault
            bf16x8 blo = *(const bf16x8*)(Wrow[t] + kB);
            bf16x8 bhi = *(const bf16x8*)(Wrow[t] + kB + 16);
            b[t] = __builtin_shufflevector(blo, bhi,
                0,1,2,3,4,5,6,7,8,9,10,11,12,13,14,15);
        }

        // 8 independent WMMAs back-to-back
        #pragma unroll
        for (int t = 0; t < 4; ++t) {
            acc[0][t] = __builtin_amdgcn_wmma_f32_16x16x32_bf16(
                false, a0, false, b[t], (short)0, acc[0][t], false, false);
            acc[1][t] = __builtin_amdgcn_wmma_f32_16x16x32_bf16(
                false, a1, false, b[t], (short)0, acc[1][t], false, false);
        }
    }

    // -------- peeled K tail (only dt_proj: K=48) --------
    if (Kfull < K) {
        const int kk = Kfull;
        const int kA = kk + half * 8;
        const int kB = kk + half * 8;

        bf16x16 a0, a1, b[4];
        #pragma unroll
        for (int e = 0; e < 16; ++e) {
            int k = kA + (e & 7) + ((e >> 3) << 4);
            a0[e] = (k < K) ? Arow0[k] : (__bf16)0.0f;
            a1[e] = (k < K) ? Arow1[k] : (__bf16)0.0f;
        }
        #pragma unroll
        for (int t = 0; t < 4; ++t)
            #pragma unroll
            for (int e = 0; e < 16; ++e) {
                int k = kB + (e & 7) + ((e >> 3) << 4);
                b[t][e] = (k < K) ? Wrow[t][k] : (__bf16)0.0f;
            }
        #pragma unroll
        for (int t = 0; t < 4; ++t) {
            acc[0][t] = __builtin_amdgcn_wmma_f32_16x16x32_bf16(
                false, a0, false, b[t], (short)0, acc[0][t], false, false);
            acc[1][t] = __builtin_amdgcn_wmma_f32_16x16x32_bf16(
                false, a1, false, b[t], (short)0, acc[1][t], false, false);
        }
    }

    // -------- store C --------
    if (accumulate) {
        #pragma unroll
        for (int s = 0; s < 2; ++s)
            #pragma unroll
            for (int t = 0; t < 4; ++t) {
                int n = nblk + t * 16 + lan15;
                if (n >= N) continue;
                #pragma unroll
                for (int r = 0; r < 8; ++r) {
                    int m = m0 + s * 16 + r + half * 8;
                    if (m >= M) continue;
                    C[(size_t)m * ldc + n] += acc[s][t][r];
                }
            }
    } else {
        // streaming output (e.g. 412MB logits): non-temporal, keep L2 for weights
        #pragma unroll
        for (int s = 0; s < 2; ++s)
            #pragma unroll
            for (int t = 0; t < 4; ++t) {
                int n = nblk + t * 16 + lan15;
                if (n >= N) continue;
                #pragma unroll
                for (int r = 0; r < 8; ++r) {
                    int m = m0 + s * 16 + r + half * 8;
                    if (m >= M) continue;
                    __builtin_nontemporal_store(acc[s][t][r], &C[(size_t)m * ldc + n]);
                }
            }
    }
}

// ---------------------------------------------------------------------------
// fp32 -> bf16 bulk conversion, 4 elements per thread (vectorized b128/b64)
__global__ void cvt_bf16_kernel(const float* __restrict__ in,
                                __bf16* __restrict__ out, long n4)
{
    long i = (long)blockIdx.x * blockDim.x + threadIdx.x;
    if (i >= n4) return;
    float4 v = ((const float4*)in)[i];
    bf16x4 o;
    o[0] = (__bf16)v.x; o[1] = (__bf16)v.y;
    o[2] = (__bf16)v.z; o[3] = (__bf16)v.w;
    ((bf16x4*)out)[i] = o;
}

__global__ void embed_kernel(const int* __restrict__ ids,
                             const float* __restrict__ emb,
                             float* __restrict__ x, int total)
{
    int i = blockIdx.x * blockDim.x + threadIdx.x;
    if (i >= total) return;
    int t = i / DMODEL, d = i % DMODEL;
    x[i] = emb[(size_t)ids[t] * DMODEL + d];
}

// rmsnorm over fp32 x, bf16 output (feeds GEMM A operand directly)
__global__ __launch_bounds__(256) void rmsnorm_kernel(
    const float* __restrict__ x, const float* __restrict__ w,
    __bf16* __restrict__ out, int D)
{
    __shared__ float red[8];
    const int t = blockIdx.x;
    const float* xr = x + (size_t)t * D;
    float s = 0.0f;
    for (int i = threadIdx.x; i < D; i += 256) { float v = xr[i]; s += v * v; }
    #pragma unroll
    for (int off = 16; off > 0; off >>= 1) s += __shfl_xor(s, off, 32);
    if ((threadIdx.x & 31) == 0) red[threadIdx.x >> 5] = s;
    __syncthreads();
    if (threadIdx.x < 8) {
        s = red[threadIdx.x];
        #pragma unroll
        for (int off = 4; off > 0; off >>= 1) s += __shfl_xor(s, off, 8);
        if (threadIdx.x == 0) red[0] = s;
    }
    __syncthreads();
    float inv = rsqrtf(red[0] / (float)D + 1e-5f);
    for (int i = threadIdx.x; i < D; i += 256)
        out[(size_t)t * D + i] = (__bf16)(xr[i] * inv * w[i]);
}

// depthwise causal conv (k=4) over first DINNER cols of xz, SiLU, bf16 out
__global__ void conv_silu_kernel(const float* __restrict__ xz,
                                 const float* __restrict__ cw,
                                 const float* __restrict__ cb,
                                 __bf16* __restrict__ xi, int total)
{
    int i = blockIdx.x * blockDim.x + threadIdx.x;
    if (i >= total) return;
    int t = i / DINNER, d = i % DINNER;
    float s = cb[d];
    #pragma unroll
    for (int j = 0; j < DCONV; ++j) {
        int tt = t - (DCONV - 1) + j;
        if (tt >= 0) s += cw[d * DCONV + j] * xz[(size_t)tt * (2 * DINNER) + d];
    }
    xi[i] = (__bf16)(s / (1.0f + __expf(-s)));   // silu
}

__global__ void dt_softplus_kernel(float* __restrict__ delta,
                                   const float* __restrict__ b, int total)
{
    int i = blockIdx.x * blockDim.x + threadIdx.x;
    if (i >= total) return;
    int d = i % DINNER;
    float v = delta[i] + b[d];
    delta[i] = (v > 20.0f) ? v : log1pf(__expf(v));
}

// selective scan: 16 lanes per channel (one per state), 16 channels per block
__global__ __launch_bounds__(256) void scan_kernel(
    const __bf16* __restrict__ xi,    // u   [L, DINNER] (bf16)
    const float* __restrict__ delta,  //     [L, DINNER]
    const float* __restrict__ xdbl,   //     [L, 80]  (B at +48, C at +64)
    const float* __restrict__ A_log,  //     [DINNER, DSTATE]
    const float* __restrict__ Dp,     //     [DINNER]
    const float* __restrict__ xz,     // res at col DINNER+d, row stride 2*DINNER
    __bf16* __restrict__ y, int L)    // bf16 out (feeds out_proj GEMM A)
{
    const int d = blockIdx.x * 16 + (threadIdx.x >> 4);
    const int n = threadIdx.x & 15;
    const float Aval = -__expf(A_log[d * DSTATE + n]);
    const float Dval = Dp[d];
    float h = 0.0f;
    for (int t = 0; t < L; ++t) {
        const float* row = xdbl + (size_t)t * XDBL_W;
        float dt = delta[(size_t)t * DINNER + d];
        float u  = (float)xi[(size_t)t * DINNER + d];
        float Bt = row[DTRANK + n];
        float Ct = row[DTRANK + DSTATE + n];
        h = __expf(dt * Aval) * h + dt * Bt * u;
        float acc = h * Ct;
        acc += __shfl_xor(acc, 1, 16);
        acc += __shfl_xor(acc, 2, 16);
        acc += __shfl_xor(acc, 4, 16);
        acc += __shfl_xor(acc, 8, 16);
        if (n == 0) {
            float res = xz[(size_t)t * (2 * DINNER) + DINNER + d];
            float yv = acc + u * Dval;
            y[(size_t)t * DINNER + d] = (__bf16)(yv * (res / (1.0f + __expf(-res))));
        }
    }
}

// ---------------------------------------------------------------------------
extern "C" void kernel_launch(void* const* d_in, const int* in_sizes, int n_in,
                              void* d_out, int out_size, void* d_ws, size_t ws_size,
                              hipStream_t stream)
{
    const int*   ids      = (const int*)d_in[0];
    const float* emb      = (const float*)d_in[1];
    const float* in_w     = (const float*)d_in[2];
    const float* conv_w   = (const float*)d_in[3];
    const float* conv_b   = (const float*)d_in[4];
    const float* xp_w     = (const float*)d_in[5];
    const float* dtp_w    = (const float*)d_in[6];
    const float* dtp_b    = (const float*)d_in[7];
    const float* A_log    = (const float*)d_in[8];
    const float* Dp       = (const float*)d_in[9];
    const float* out_w    = (const float*)d_in[10];
    const float* norm_w   = (const float*)d_in[11];
    const float* norm_f_w = (const float*)d_in[12];

    // ---- workspace carve-up (64B-aligned regions) ----
    char* base = (char*)d_ws;
    size_t off = 0;
    auto alloc = [&](size_t bytes) -> char* {
        char* p = base + off;
        off = (off + bytes + 63) & ~(size_t)63;
        return p;
    };
    // fp32 activations
    float* x     = (float*)alloc(sizeof(float) * L_SEQ * DMODEL);
    float* xz    = (float*)alloc(sizeof(float) * L_SEQ * 2 * DINNER);
    float* xdbl  = (float*)alloc(sizeof(float) * L_SEQ * XDBL_W);
    float* delta = (float*)alloc(sizeof(float) * L_SEQ * DINNER);
    // bf16 activations
    __bf16* h_bf    = (__bf16*)alloc(sizeof(__bf16) * L_SEQ * DMODEL);
    __bf16* xi_bf   = (__bf16*)alloc(sizeof(__bf16) * L_SEQ * DINNER);
    __bf16* xdbl_bf = (__bf16*)alloc(sizeof(__bf16) * L_SEQ * XDBL_W);
    __bf16* y_bf    = (__bf16*)alloc(sizeof(__bf16) * L_SEQ * DINNER);
    // bf16 weights (converted once per call)
    __bf16* emb_bf  = (__bf16*)alloc(sizeof(__bf16) * (size_t)NVOCAB * DMODEL);
    __bf16* inw_bf  = (__bf16*)alloc(sizeof(__bf16) * (size_t)NLAYER * 2 * DINNER * DMODEL);
    __bf16* xpw_bf  = (__bf16*)alloc(sizeof(__bf16) * (size_t)NLAYER * XDBL_W * DINNER);
    __bf16* dtw_bf  = (__bf16*)alloc(sizeof(__bf16) * (size_t)NLAYER * DINNER * DTRANK);
    __bf16* outw_bf = (__bf16*)alloc(sizeof(__bf16) * (size_t)NLAYER * DMODEL * DINNER);

    const dim3 blk128(128), blk256(256);
    const int MBLK = 128;
    auto cvt = [&](const float* src, __bf16* dst, size_t n) {
        long n4 = (long)(n / 4);
        cvt_bf16_kernel<<<(n4 + 255) / 256, blk256, 0, stream>>>(src, dst, n4);
    };

    // ---- one-time conversions ----
    cvt(emb,   emb_bf,  (size_t)NVOCAB * DMODEL);
    cvt(in_w,  inw_bf,  (size_t)NLAYER * 2 * DINNER * DMODEL);
    cvt(xp_w,  xpw_bf,  (size_t)NLAYER * XDBL_W * DINNER);
    cvt(dtp_w, dtw_bf,  (size_t)NLAYER * DINNER * DTRANK);
    cvt(out_w, outw_bf, (size_t)NLAYER * DMODEL * DINNER);

    embed_kernel<<<(L_SEQ * DMODEL + 255) / 256, blk256, 0, stream>>>(
        ids, emb, x, L_SEQ * DMODEL);

    for (int i = 0; i < NLAYER; ++i) {
        const __bf16* inw = inw_bf  + (size_t)i * (2 * DINNER) * DMODEL;
        const __bf16* xpw = xpw_bf  + (size_t)i * XDBL_W * DINNER;
        const __bf16* dtw = dtw_bf  + (size_t)i * DINNER * DTRANK;
        const __bf16* ow  = outw_bf + (size_t)i * DMODEL * DINNER;
        const float*  cw  = conv_w + (size_t)i * DINNER * DCONV;
        const float*  cb  = conv_b + (size_t)i * DINNER;
        const float*  dtb = dtp_b  + (size_t)i * DINNER;
        const float*  Al  = A_log  + (size_t)i * DINNER * DSTATE;
        const float*  Dv  = Dp     + (size_t)i * DINNER;
        const float*  nw  = norm_w + (size_t)i * DMODEL;

        rmsnorm_kernel<<<L_SEQ, blk256, 0, stream>>>(x, nw, h_bf, DMODEL);

        // xz = h @ in_w^T   [2048, 3072]
        gemm_bf16_wmma<<<dim3(L_SEQ / MBLK, (2 * DINNER + 63) / 64), blk128, 0, stream>>>(
            h_bf, DMODEL, inw, DMODEL, xz, 2 * DINNER, L_SEQ, 2 * DINNER, DMODEL, 0);

        conv_silu_kernel<<<(L_SEQ * DINNER + 255) / 256, blk256, 0, stream>>>(
            xz, cw, cb, xi_bf, L_SEQ * DINNER);

        // x_dbl = xi @ xp_w^T   [2048, 80]
        gemm_bf16_wmma<<<dim3(L_SEQ / MBLK, (XDBL_W + 63) / 64), blk128, 0, stream>>>(
            xi_bf, DINNER, xpw, DINNER, xdbl, XDBL_W, L_SEQ, XDBL_W, DINNER, 0);

        cvt(xdbl, xdbl_bf, (size_t)L_SEQ * XDBL_W);

        // delta_lin = x_dbl[:, :48] @ dt_proj_w^T   [2048, 1536]
        gemm_bf16_wmma<<<dim3(L_SEQ / MBLK, (DINNER + 63) / 64), blk128, 0, stream>>>(
            xdbl_bf, XDBL_W, dtw, DTRANK, delta, DINNER, L_SEQ, DINNER, DTRANK, 0);

        dt_softplus_kernel<<<(L_SEQ * DINNER + 255) / 256, blk256, 0, stream>>>(
            delta, dtb, L_SEQ * DINNER);

        scan_kernel<<<DINNER / 16, blk256, 0, stream>>>(
            xi_bf, delta, xdbl, Al, Dv, xz, y_bf, L_SEQ);

        // x += y @ out_w^T   [2048, 768]  (residual via accumulate)
        gemm_bf16_wmma<<<dim3(L_SEQ / MBLK, (DMODEL + 63) / 64), blk128, 0, stream>>>(
            y_bf, DINNER, ow, DINNER, x, DMODEL, L_SEQ, DMODEL, DINNER, 1);
    }

    rmsnorm_kernel<<<L_SEQ, blk256, 0, stream>>>(x, norm_f_w, h_bf, DMODEL);

    // logits = h @ emb^T   [2048, 50264]
    gemm_bf16_wmma<<<dim3(L_SEQ / MBLK, (NVOCAB + 63) / 64), blk128, 0, stream>>>(
        h_bf, DMODEL, emb_bf, DMODEL, (float*)d_out, NVOCAB, L_SEQ, NVOCAB, DMODEL, 0);
}
